// DSQGAttentionV10_9483287789631
// MI455X (gfx1250) — matrix operand order repaired and checked by reading, not compiled, not run on hardware
//
#include <hip/hip_runtime.h>

typedef __attribute__((ext_vector_type(16))) _Float16 v16h;
typedef __attribute__((ext_vector_type(8)))  _Float16 v8h;
typedef __attribute__((ext_vector_type(8)))  float    v8f;
typedef __attribute__((ext_vector_type(4)))  unsigned int u32x4;
typedef __attribute__((ext_vector_type(8)))  int      i32x8;
typedef __attribute__((ext_vector_type(4)))  int      i32x4;

#define B_  8
#define H_  16
#define N_  8192
#define HD_ 64
#define J_  20
#define NTILES (N_/16)   // 512

// LDS arena layout (byte offsets; static LDS allocation starts at 0,
// so these double as TDM D#.lds_addr values)
#define OFF_VST 0          // f32 V stage: 8 blocks * 16*64*4      = 32768
#define OFF_LVT 32768      // f16 V transposed: [4 pr][64 c][32 kw] = 16384
#define OFF_LS  49152      // f32 scores: [8 t][16 r][16 c]         =  8192
#define OFF_LT  57344      // f32 T=Q*SE^T: [16 r][32 j]            =  2048
#define OFF_LP  59392      // f16 P: [4 pr][16 r][32 col32]         =  4096
#define LDS_TOTAL 63488

__constant__ int c_off[J_] = {1,2,3,4,5,6,7,8,9,11,13,15,16,23,32,64,128,256,512,1024};
__constant__ int c_blk[8]  = {0,16,32,64,128,256,512,1024};
// block-selection tables: t = c_tlo[j] + (m < c_thr[j]); col = m - c_thr[j] (+16 if hi)
__constant__ int c_thr[J_] = {1,2,3,4,5,6,7,8,9,11,13,15, 0, 7, 0,0,0,0,0,0};
__constant__ int c_tlo[J_] = {0,0,0,0,0,0,0,0,0, 0, 0, 0, 1, 1, 2,3,4,5,6,7};

// A-fragment (16-bit, 16x32) element->K mapping: contiguous k = 16*hs + i
__device__ __forceinline__ int a_kmap(int i, int hs) {
    return i + (hs << 4);
}

__device__ __forceinline__ v16h cat8(v8h lo, v8h hi) {
    return __builtin_shufflevector(lo, hi, 0,1,2,3,4,5,6,7,8,9,10,11,12,13,14,15);
}

__global__ __launch_bounds__(32)
void dsqg_attn_wmma(const float* __restrict__ q, const float* __restrict__ k,
                    const float* __restrict__ v, const float* __restrict__ pb,
                    const float* __restrict__ se, float* __restrict__ out)
{
    __shared__ __align__(16) unsigned char smem[LDS_TOTAL];
    float*    vst  = (float*)(smem + OFF_VST);
    _Float16* lvt  = (_Float16*)(smem + OFF_LVT);
    float*    ls   = (float*)(smem + OFF_LS);
    float*    lt   = (float*)(smem + OFF_LT);
    _Float16* lp16 = (_Float16*)(smem + OFF_LP);

    const int lane = threadIdx.x;     // 0..31, wave32
    const int m    = lane & 15;
    const int hs   = lane >> 4;

    const int tile = blockIdx.x & (NTILES - 1);
    const int bh   = blockIdx.x >> 9;
    const int h    = bh & (H_ - 1);
    const int n0   = tile << 4;
    const size_t base = (size_t)bh * N_ * HD_;

    const float sc = 0.125f;  // 1/sqrt(64)

    // ============ issue 8 TDM tile loads: V blocks -> LDS (f32) ============
    #pragma unroll
    for (int t = 0; t < 8; ++t) {
        int rb = n0 - c_blk[t];
        if (rb < 0) rb = 0;   // safe: blocks with negative base are never used by valid entries
        unsigned long long ga =
            (unsigned long long)(const void*)(v + base + (size_t)rb * HD_);
        u32x4 g0;
        g0[0] = 1u;                                   // count=1 (valid user D#)
        g0[1] = (unsigned)(OFF_VST + t * 4096);       // lds_addr
        g0[2] = (unsigned)ga;                         // global_addr[31:0]
        g0[3] = (unsigned)(ga >> 32) | (2u << 30);    // global_addr[56:32] | type=2
        i32x8 g1;
        g1[0] = (int)(2u << 16);    // workgroup_mask=0, data_size=2 (4 bytes)
        g1[1] = (int)(64u << 16);   // tensor_dim0 = 64 elements
        g1[2] = (int)(16u << 16);   // tensor_dim1 = 16 rows
        g1[3] = (int)(64u << 16);   // tile_dim0 = 64
        g1[4] = 16;                 // tile_dim1 = 16, tile_dim2 = 0
        g1[5] = 64;                 // tensor_dim0_stride = 64
        g1[6] = 0;
        g1[7] = 0;
        i32x4 z4; z4[0] = 0; z4[1] = 0; z4[2] = 0; z4[3] = 0;
#if __clang_major__ >= 23
        i32x8 z8; z8[0]=0; z8[1]=0; z8[2]=0; z8[3]=0; z8[4]=0; z8[5]=0; z8[6]=0; z8[7]=0;
        __builtin_amdgcn_tensor_load_to_lds(g0, g1, z4, z4, z8, 0);
#else
        __builtin_amdgcn_tensor_load_to_lds(g0, g1, z4, z4, 0);
#endif
    }

    // ============ Q A-fragments (two K=32 chunks) ============
    v16h qa0, qa1;
    {
        const float* qrow = q + base + (size_t)(n0 + m) * HD_;
        #pragma unroll
        for (int i = 0; i < 16; ++i) {
            int kk = a_kmap(i, hs);
            qa0[i] = (_Float16)qrow[kk];
            qa1[i] = (_Float16)qrow[32 + kk];
        }
    }

    // ============ scores: S_t = Q * K_t^T (WMMA) ============
    #pragma unroll
    for (int t = 0; t < 8; ++t) {
        int rb = n0 - c_blk[t];
        if (rb < 0) rb = 0;  // safe clamp (see above)
        const float* kr = k + base + (size_t)(rb + m) * HD_ + (hs << 4);
        v16h b0, b1;
        #pragma unroll
        for (int i = 0; i < 16; ++i) {
            b0[i] = (_Float16)kr[i];
            b1[i] = (_Float16)kr[32 + i];
        }
        v8f acc = {};
        acc = __builtin_amdgcn_wmma_f32_16x16x32_f16(false, qa0, false, b0, (short)0, acc, false, false);
        acc = __builtin_amdgcn_wmma_f32_16x16x32_f16(false, qa1, false, b1, (short)0, acc, false, false);
        #pragma unroll
        for (int vv = 0; vv < 8; ++vv)
            ls[(t * 16 + vv + (hs << 3)) * 16 + m] = acc[vv];
    }

    // ============ T = Q * SE^T (16 x 20) ============
    #pragma unroll
    for (int nb = 0; nb < 2; ++nb) {
        int j = (nb << 4) + m;
        v16h b0, b1;
        if (j < J_) {
            const float* sr = se + j * HD_ + (hs << 4);
            #pragma unroll
            for (int i = 0; i < 16; ++i) {
                b0[i] = (_Float16)sr[i];
                b1[i] = (_Float16)sr[32 + i];
            }
        } else {
            #pragma unroll
            for (int i = 0; i < 16; ++i) { b0[i] = (_Float16)0.0f; b1[i] = (_Float16)0.0f; }
        }
        v8f acc = {};
        acc = __builtin_amdgcn_wmma_f32_16x16x32_f16(false, qa0, false, b0, (short)0, acc, false, false);
        acc = __builtin_amdgcn_wmma_f32_16x16x32_f16(false, qa1, false, b1, (short)0, acc, false, false);
        #pragma unroll
        for (int vv = 0; vv < 8; ++vv)
            lt[(vv + (hs << 3)) * 32 + (nb << 4) + m] = acc[vv];
    }

    // ============ zero P region (1024 dwords) ============
    {
        unsigned* lpz = (unsigned*)lp16;
        #pragma unroll
        for (int i = 0; i < 32; ++i) lpz[(i << 5) + lane] = 0u;
    }

    // ============ V: f32 stage -> f16 transposed [pr][col][kw] ============
    __builtin_amdgcn_s_wait_tensorcnt(0);
    #pragma unroll
    for (int t = 0; t < 8; ++t) {
        int pr = t >> 1, hf = t & 1;
        #pragma unroll
        for (int it = 0; it < 16; ++it) {
            int p  = (it << 5) + lane;        // 0..511 over (row-pair, col)
            int r2 = p & 7;                   // row pair -> rows 2*r2, 2*r2+1
            int c  = p >> 3;                  // 0..63
            float f0 = vst[(t * 16 + (r2 << 1)) * 64 + c];
            float f1 = vst[(t * 16 + (r2 << 1) + 1) * 64 + c];
            union { unsigned u; _Float16 hh[2]; } pku;
            pku.hh[0] = (_Float16)f0;
            pku.hh[1] = (_Float16)f1;
            *(unsigned*)(lvt + ((pr * 64 + c) * 32 + (hf << 4) + (r2 << 1))) = pku.u;
        }
    }
    __syncthreads();

    // ============ per-row softmax over 20 offsets ============
    if (lane < 16) {
        int n = n0 + lane;
        float sj[J_];
        int   idx[J_];           // f16 P-store index, or -1 if invalid
        float mx = -3.4e38f;
        #pragma unroll
        for (int j = 0; j < J_; ++j) {
            int d = c_off[j];
            idx[j] = -1;
            if (n >= d) {
                int thr = c_thr[j];
                int lo  = (lane >= thr) ? 1 : 0;
                int t   = c_tlo[j] + 1 - lo;
                int col = lane - thr + (lo ? 0 : 16);
                float sv = (ls[(t * 16 + lane) * 16 + col] + lt[lane * 32 + j]) * sc
                         + pb[j * H_ + h];
                sj[j] = sv;
                // P layout: [pr][row][ (t&1)*16 + col ]
                idx[j] = ((t >> 1) * 16 + lane) * 32 + ((t & 1) << 4) + col;
                mx = fmaxf(mx, sv);
            }
        }
        float den = 0.0f;
        #pragma unroll
        for (int j = 0; j < J_; ++j)
            if (idx[j] >= 0) { sj[j] = __expf(sj[j] - mx); den += sj[j]; }
        float inv = (den > 0.0f) ? (1.0f / den) : 0.0f;
        #pragma unroll
        for (int j = 0; j < J_; ++j)
            if (idx[j] >= 0) lp16[idx[j]] = (_Float16)(sj[j] * inv);
    }
    __syncthreads();

    // ============ out = sum_t P_t * V_t (pairs, fragment-native LDS) ============
    #pragma unroll
    for (int c = 0; c < 4; ++c) {
        v8f acc = {};
        #pragma unroll
        for (int pr = 0; pr < 4; ++pr) {
            // A fragment: contiguous f16 run lp16[pr][m][hs*16 .. +15]
            const _Float16* ap = lp16 + (pr * 16 + m) * 32;
            v16h a = cat8(*(const v8h*)(ap + (hs << 4)),
                          *(const v8h*)(ap + (hs << 4) + 8));
            // B fragment: contiguous f16 run lvt[pr][c*16+m][hs*16 .. +15]
            const _Float16* bp = lvt + (pr * 64 + (c << 4) + m) * 32 + (hs << 4);
            v16h b = cat8(*(const v8h*)bp, *(const v8h*)(bp + 8));
            acc = __builtin_amdgcn_wmma_f32_16x16x32_f16(false, a, false, b, (short)0, acc, false, false);
        }
        #pragma unroll
        for (int vv = 0; vv < 8; ++vv)
            out[base + (size_t)(n0 + vv + (hs << 3)) * HD_ + (c << 4) + m] = acc[vv];
    }
}

extern "C" void kernel_launch(void* const* d_in, const int* in_sizes, int n_in,
                              void* d_out, int out_size, void* d_ws, size_t ws_size,
                              hipStream_t stream) {
    (void)in_sizes; (void)n_in; (void)out_size; (void)d_ws; (void)ws_size;
    const float* q  = (const float*)d_in[0];
    const float* k  = (const float*)d_in[1];
    const float* v  = (const float*)d_in[2];
    const float* pb = (const float*)d_in[3];
    const float* se = (const float*)d_in[4];
    float* out = (float*)d_out;

    dim3 grid(B_ * H_ * NTILES);   // 65536 tiles of 16 queries
    dim3 block(32);                // one wave32 per workgroup
    dsqg_attn_wmma<<<grid, block, 0, stream>>>(q, k, v, pb, se, out);
}
